// MultiHeadAttention_72550587564683
// MI455X (gfx1250) — compile-verified, matrix operand order checked by reading, not run on hardware
//
#include <hip/hip_runtime.h>
#include <hip/hip_bf16.h>

typedef __bf16 bf16_t;
typedef __bf16 v16bf __attribute__((ext_vector_type(16)));
typedef float  v8f   __attribute__((ext_vector_type(8)));
typedef int    v4i   __attribute__((ext_vector_type(4)));

union Frag {
  v16bf v;
  uint4 u[2];
};

constexpr int Ss   = 4096;
constexpr int Dd   = 1024;
constexpr int Mtot = 4 * Ss;  // 16384 rows total (B*S)

// ---- CDNA5 async global->LDS copy (ASYNCcnt-tracked, no VGPR round trip) ----
#if defined(__has_builtin)
#if __has_builtin(__builtin_amdgcn_global_load_async_to_lds_b128) && \
    __has_builtin(__builtin_amdgcn_s_wait_asynccnt)
#define HAVE_ASYNC 1
#endif
#endif
#ifndef HAVE_ASYNC
#define HAVE_ASYNC 0
#endif

#if HAVE_ASYNC
typedef __attribute__((address_space(1))) v4i* gas_v4i;
typedef __attribute__((address_space(3))) v4i* las_v4i;
__device__ __forceinline__ void cp_async16(const void* g, void* l) {
  __builtin_amdgcn_global_load_async_to_lds_b128(
      (gas_v4i)g, (las_v4i)l, 0, 0);
}
__device__ __forceinline__ void cp_async_wait() {
  __builtin_amdgcn_s_wait_asynccnt(0);
}
#endif

// =====================================================================
// GEMM: C[M,1024] = A[M,1024] * B[1024,1024]   via v_wmma_f32_16x16x32_bf16
//   Block: 256 threads (8 waves), tile 128(M) x 64(N), K-step 32.
//   A_F32:    A is fp32 (converted to bf16 while staging to LDS), else bf16
//             (bf16 path uses async global->LDS when available).
//   OUT_F32:  write fp32 (+bias); else write bf16.
//   TRANS_OUT: write bf16 transposed per-batch: Ct[b][n][s]  (for V^T).
// =====================================================================
template<bool A_F32, bool OUT_F32, bool TRANS_OUT>
__global__ __launch_bounds__(256) void gemm_wmma(
    const void* __restrict__ Ap, const float* __restrict__ Bw,
    void* __restrict__ Cp, const float* __restrict__ bias, float alpha)
{
  constexpr int Kdim = 1024;
  constexpr int LDA  = 1024;
  constexpr int LDB  = 1024;
  constexpr int AST  = 40;            // LDS row stride in bf16 elems (80B, 16B-aligned)
  __shared__ bf16_t As[128 * AST];    // 128 x 32 A tile (bf16)
  __shared__ bf16_t Bs[64 * AST];     // 64(N) x 32(K) B tile, stored transposed

  const int tid  = threadIdx.x;
  const int lane = tid & 31, w = tid >> 5;
  const int half = lane >> 4, l16 = lane & 15;

  const int m0 = blockIdx.x * 128;
  const int n0 = blockIdx.y * 64;

  v8f acc[4];
  for (int i = 0; i < 4; ++i)
    for (int j = 0; j < 8; ++j) acc[i][j] = 0.f;

  const int arow = tid >> 1;        // 0..127  (2 threads per A row)
  const int akh  = (tid & 1) * 16;  // K half: 0 or 16
  const int bkr  = tid >> 3;        // 0..31   (8 threads per B k-row)
  const int bnc  = (tid & 7) * 8;   // 0..56

  for (int k0 = 0; k0 < Kdim; k0 += 32) {
    // ---- cooperative A tile -> LDS (bf16) ----
    if (A_F32) {
      bf16_t tmp[16];
      const float* src = (const float*)Ap + (long)(m0 + arow) * LDA + k0 + akh;
      for (int i = 0; i < 4; ++i) {
        float4 f = *(const float4*)(src + i * 4);
        tmp[i*4+0] = (bf16_t)f.x; tmp[i*4+1] = (bf16_t)f.y;
        tmp[i*4+2] = (bf16_t)f.z; tmp[i*4+3] = (bf16_t)f.w;
      }
      *(uint4*)&As[arow * AST + akh]     = *(uint4*)&tmp[0];
      *(uint4*)&As[arow * AST + akh + 8] = *(uint4*)&tmp[8];
    } else {
      const bf16_t* src = (const bf16_t*)Ap + (long)(m0 + arow) * LDA + k0 + akh;
      bf16_t* dst = &As[arow * AST + akh];
#if HAVE_ASYNC
      cp_async16(src, dst);
      cp_async16(src + 8, dst + 8);
#else
      bf16_t tmp[16];
      *(uint4*)&tmp[0] = *(const uint4*)src;
      *(uint4*)&tmp[8] = *(const uint4*)(src + 8);
      *(uint4*)&dst[0] = *(uint4*)&tmp[0];
      *(uint4*)&dst[8] = *(uint4*)&tmp[8];
#endif
    }
    // ---- cooperative B tile -> LDS, transposed so B-fragments are contiguous ----
    {
      const float* src = Bw + (long)(k0 + bkr) * LDB + n0 + bnc;
      float4 f0 = *(const float4*)src;
      float4 f1 = *(const float4*)(src + 4);
      bf16_t t[8] = {(bf16_t)f0.x,(bf16_t)f0.y,(bf16_t)f0.z,(bf16_t)f0.w,
                     (bf16_t)f1.x,(bf16_t)f1.y,(bf16_t)f1.z,(bf16_t)f1.w};
      for (int i = 0; i < 8; ++i) Bs[(bnc + i) * AST + bkr] = t[i];
    }
#if HAVE_ASYNC
    if (!A_F32) cp_async_wait();
#endif
    __syncthreads();

    // ---- fragments per ISA layout: lanes 0-15 K=0..7,16..23 ; lanes 16-31 K=8..15,24..31 ----
    Frag a;
    a.u[0] = *(const uint4*)&As[(w*16 + l16) * AST + half*8];
    a.u[1] = *(const uint4*)&As[(w*16 + l16) * AST + 16 + half*8];
    for (int ns = 0; ns < 4; ++ns) {
      Frag b;
      b.u[0] = *(const uint4*)&Bs[(ns*16 + l16) * AST + half*8];
      b.u[1] = *(const uint4*)&Bs[(ns*16 + l16) * AST + 16 + half*8];
      acc[ns] = __builtin_amdgcn_wmma_f32_16x16x32_bf16(
          false, a.v, false, b.v, (short)0, acc[ns], false, false);
    }
    __syncthreads();
  }

  // ---- epilogue: C/D layout = VGPR r -> rows r (lanes 0-15) / r+8 (lanes 16-31) ----
  for (int ns = 0; ns < 4; ++ns) {
    for (int r = 0; r < 8; ++r) {
      int gm = m0 + w*16 + r + half*8;
      int gn = n0 + ns*16 + l16;
      float vv = acc[ns][r] * alpha;
      if (bias) vv += bias[gn];
      if (OUT_F32) {
        ((float*)Cp)[(long)gm * 1024 + gn] = vv;
      } else if (!TRANS_OUT) {
        ((bf16_t*)Cp)[(long)gm * 1024 + gn] = (bf16_t)vv;
      } else {
        int bb = gm >> 12, s = gm & (Ss - 1);
        ((bf16_t*)Cp)[((long)bb * Dd + gn) * Ss + s] = (bf16_t)vv;
      }
    }
  }
}

// =====================================================================
// Fused attention: one 256-thread workgroup per (batch, 16-query-row tile).
// 296KB of CDNA5's 320KB LDS: Q tile (bf16) + full 16x4096 score block (f32).
// =====================================================================
constexpr int    QSTR     = 1032;  // bf16 elems per Q row in LDS (2064B, 16B aligned)
constexpr int    SSTR     = 4104;  // f32 elems per score row in LDS (16416B, 16B aligned)
constexpr size_t ATTN_LDS = (size_t)16 * QSTR * sizeof(bf16_t)
                          + (size_t)16 * SSTR * sizeof(float);   // 295,680 B

__global__ __launch_bounds__(256) void attn_fused(
    const bf16_t* __restrict__ Q, const bf16_t* __restrict__ Kb,
    const bf16_t* __restrict__ Vt, bf16_t* __restrict__ O)
{
  extern __shared__ char smem[];
  bf16_t* Qs = (bf16_t*)smem;
  float*  Sc = (float*)(smem + 16 * QSTR * sizeof(bf16_t));

  const int tid  = threadIdx.x;
  const int lane = tid & 31, w = tid >> 5;
  const int half = lane >> 4, l16 = lane & 15;
  const int qt = blockIdx.x;  // query tile (16 rows)
  const int bb = blockIdx.y;  // batch

  // ---- phase 1: Q tile -> LDS (16 x 1024 bf16, 32KB), async when available ----
  {
    int r  = tid >> 4;
    int cb = (tid & 15) * 64;
    const bf16_t* src = Q + ((long)(bb * Ss + qt * 16 + r)) * Dd + cb;
    bf16_t* dst = &Qs[r * QSTR + cb];
#if HAVE_ASYNC
    for (int i = 0; i < 8; ++i) cp_async16(src + i * 8, dst + i * 8);
    cp_async_wait();
#else
    for (int i = 0; i < 8; ++i)
      *(uint4*)(dst + i * 8) = *(const uint4*)(src + i * 8);
#endif
  }
  __syncthreads();

  // ---- phase 2: scores = Q*K^T (scale pre-folded into Q). Waves split key tiles.
  //      B-fragment column n of K^T == K row n -> contiguous global b128 loads.
  //      Prefetch next key tile's rows while this tile's 32 WMMAs run. ----
  for (int j = w; j < Ss / 16; j += 8) {
    v8f s;
    for (int i = 0; i < 8; ++i) s[i] = 0.f;
    const bf16_t* Krow = Kb + ((long)(bb * Ss + j * 16 + l16)) * Dd;
    const bf16_t* Knext = (j + 8 < Ss / 16)
        ? Kb + ((long)(bb * Ss + (j + 8) * 16 + l16)) * Dd : Krow;
    for (int kd = 0; kd < Dd; kd += 32) {
      __builtin_prefetch(Knext + kd, 0, 3);   // global_prefetch_b8
      Frag a, b;
      a.u[0] = *(const uint4*)&Qs[l16 * QSTR + kd + half*8];
      a.u[1] = *(const uint4*)&Qs[l16 * QSTR + kd + 16 + half*8];
      b.u[0] = *(const uint4*)(Krow + kd + half*8);
      b.u[1] = *(const uint4*)(Krow + kd + 16 + half*8);
      s = __builtin_amdgcn_wmma_f32_16x16x32_bf16(
          false, a.v, false, b.v, (short)0, s, false, false);
    }
    for (int r = 0; r < 8; ++r)
      Sc[(r + half*8) * SSTR + j*16 + l16] = s[r];
  }
  __syncthreads();

  // ---- phase 3: row softmax in LDS (16 threads per row, shfl width-16 reductions) ----
  {
    int row = tid >> 4, sub = tid & 15;
    float* Sr = Sc + row * SSTR;
    float mx = -3.0e38f;
    for (int c = sub; c < Ss; c += 16) mx = fmaxf(mx, Sr[c]);
    for (int off = 8; off; off >>= 1) mx = fmaxf(mx, __shfl_xor(mx, off, 16));
    float sum = 0.f;
    for (int c = sub; c < Ss; c += 16) { float p = __expf(Sr[c] - mx); Sr[c] = p; sum += p; }
    for (int off = 8; off; off >>= 1) sum += __shfl_xor(sum, off, 16);
    float inv = 1.f / sum;
    for (int c = sub; c < Ss; c += 16) Sr[c] *= inv;
  }
  __syncthreads();

  // ---- phase 4: O = P * V. Wave w owns output cols [128w, 128w+128).
  //      A from LDS scores (f32 -> bf16 on the fly); B rows contiguous from V^T. ----
  v8f oacc[8];
  for (int i = 0; i < 8; ++i)
    for (int j2 = 0; j2 < 8; ++j2) oacc[i][j2] = 0.f;

  for (int k0 = 0; k0 < Ss; k0 += 32) {
    Frag a;
    {
      const float* Sr = Sc + l16 * SSTR + k0;
      float tmp[16];
      *(float4*)&tmp[0]  = *(const float4*)(Sr + half*8);
      *(float4*)&tmp[4]  = *(const float4*)(Sr + half*8 + 4);
      *(float4*)&tmp[8]  = *(const float4*)(Sr + 16 + half*8);
      *(float4*)&tmp[12] = *(const float4*)(Sr + 16 + half*8 + 4);
      for (int i = 0; i < 16; ++i) ((bf16_t*)&a)[i] = (bf16_t)tmp[i];
    }
    for (int ns = 0; ns < 8; ++ns) {
      int n = w * 128 + ns * 16 + l16;
      const bf16_t* Vrow = Vt + ((long)bb * Dd + n) * Ss + k0;
      Frag b;
      b.u[0] = *(const uint4*)(Vrow + half*8);
      b.u[1] = *(const uint4*)(Vrow + 16 + half*8);
      oacc[ns] = __builtin_amdgcn_wmma_f32_16x16x32_bf16(
          false, a.v, false, b.v, (short)0, oacc[ns], false, false);
    }
  }

  for (int ns = 0; ns < 8; ++ns)
    for (int r = 0; r < 8; ++r) {
      int gq = bb * Ss + qt * 16 + r + half*8;
      int gn = w * 128 + ns * 16 + l16;
      O[(long)gq * Dd + gn] = (bf16_t)oacc[ns][r];
    }
}

// =====================================================================
extern "C" void kernel_launch(void* const* d_in, const int* in_sizes, int n_in,
                              void* d_out, int out_size, void* d_ws, size_t ws_size,
                              hipStream_t stream) {
  (void)in_sizes; (void)n_in; (void)out_size; (void)ws_size;
  const float* x  = (const float*)d_in[0];
  const float* Wq = (const float*)d_in[1];
  const float* Wk = (const float*)d_in[2];
  const float* Wv = (const float*)d_in[3];
  const float* Wo = (const float*)d_in[4];
  const float* bo = (const float*)d_in[5];

  const size_t nElem = (size_t)Mtot * Dd;     // 16,777,216 elems (32MB bf16 each)
  bf16_t* Qb   = (bf16_t*)d_ws;
  bf16_t* Kbuf = Qb + nElem;
  bf16_t* Vt   = Kbuf + nElem;                // V stored transposed [b][D][S]
  bf16_t* Attn = Vt + nElem;

  dim3 gG(Mtot / 128, Dd / 64, 1), blk(256);
  const float scale = 0.03125f;  // 1/sqrt(1024), folded into Q

  gemm_wmma<true,  false, false><<<gG, blk, 0, stream>>>(x, Wq, Qb,   nullptr, scale);
  gemm_wmma<true,  false, false><<<gG, blk, 0, stream>>>(x, Wk, Kbuf, nullptr, 1.f);
  gemm_wmma<true,  false, true ><<<gG, blk, 0, stream>>>(x, Wv, Vt,   nullptr, 1.f);

  (void)hipFuncSetAttribute((const void*)attn_fused,
                            hipFuncAttributeMaxDynamicSharedMemorySize, (int)ATTN_LDS);
  attn_fused<<<dim3(Ss / 16, 4, 1), blk, ATTN_LDS, stream>>>(Qb, Kbuf, Vt, Attn);

  gemm_wmma<false, true,  false><<<gG, blk, 0, stream>>>(Attn, Wo, d_out, bo, 1.f);
}
